// AttentionDeform_46815143526481
// MI455X (gfx1250) — matrix operand order repaired and checked by reading, not compiled, not run on hardware
//
#include <hip/hip_runtime.h>
#include <hip/hip_bf16.h>

// ---------------------------------------------------------------------------
// AttentionDeform for MI455X (gfx1250, wave32, WMMA).
// C=128, H=4, HD=32, N=4096.  All big GEMMs + attention run on
// v_wmma_f32_16x16x32_bf16 (f32 accumulate).  Attention is fused
// flash-style (never materializes the 4x4096x4096 score matrix, which
// would be 256MB > L2).  Activations are staged point-major in bf16 so
// every WMMA B-fragment load is two contiguous b128 loads per lane, and
// all weights are pre-packed to bf16 once so A-fragment loads are also
// two b128 loads (no per-tile f32->bf16 conversion).
// ---------------------------------------------------------------------------

#define C_  128
#define N_  4096
#define H_  4
#define HD_ 32

typedef __attribute__((ext_vector_type(16))) __bf16       bf16x16;
typedef __attribute__((ext_vector_type(8)))  float        floatx8;
typedef __attribute__((ext_vector_type(4)))  unsigned int uintx4;
typedef __attribute__((ext_vector_type(8)))  unsigned int uintx8;

union BfPack { bf16x16 bf; uintx8 u; };

// Load 8+8 contiguous bf16 (two b128 loads) into a WMMA fragment register.
static __device__ __forceinline__ bf16x16 ld_bf8x2(const __bf16* p0, const __bf16* p1) {
  uintx4 a = *(const uintx4*)p0;
  uintx4 b = *(const uintx4*)p1;
  BfPack pk;
  pk.u[0] = a[0]; pk.u[1] = a[1]; pk.u[2] = a[2]; pk.u[3] = a[3];
  pk.u[4] = b[0]; pk.u[5] = b[1]; pk.u[6] = b[2]; pk.u[7] = b[3];
  return pk.bf;
}

static __device__ __forceinline__ floatx8 wmma_bf16(bf16x16 a, bf16x16 b, floatx8 c) {
  return __builtin_amdgcn_wmma_f32_16x16x32_bf16(false, a, false, b, (short)0, c,
                                                 false, false);
}

// One 16x16 output tile of D = W * B.  W is [M x CIN] bf16 row-major
// (pre-packed), B is point-major bf16 rows of stride SB ([n][SB]).
// A layout (16x32 bf16): lane<16 row=lm holds K {c0..c0+7, c0+16..c0+23};
//                        lane>=16 same row, K {c0+8..c0+15, c0+24..c0+31}.
// B layout (32x16 bf16): lane<16 col=lm holds K {c0..c0+7, c0+16..c0+23} (2/VGPR);
//                        lane>=16 same col, K {c0+8..c0+15, c0+24..c0+31}.
static __device__ __forceinline__ floatx8 gemm_tile(const __bf16* __restrict__ W, int CIN,
                                                    const __bf16* __restrict__ Bt, int SB,
                                                    int co, int n0) {
  const int l  = threadIdx.x & 31;
  const int lm = l & 15;
  const int hi = l >> 4;
  const int rowA = co * 16 + lm;
  floatx8 acc = {0.f, 0.f, 0.f, 0.f, 0.f, 0.f, 0.f, 0.f};
  for (int c0 = 0; c0 < CIN; c0 += 32) {
    const int ka = c0 + hi * 8;
    const __bf16* ap = W + rowA * CIN + ka;
    bf16x16 A = ld_bf8x2(ap, ap + 16);
    const int kb = c0 + hi * 8;
    const __bf16* bp = Bt + (n0 + lm) * SB + kb;
    bf16x16 Bm = ld_bf8x2(bp, bp + 16);
    acc = wmma_bf16(A, Bm, acc);
  }
  return acc;
}

// ---------------------------------------------------------------------------
// Kernel 0: zero the BN accumulator region (must run every launch: atomics).
__global__ void zero_stats(float* __restrict__ p) { p[threadIdx.x] = 0.f; }

// Kernel Tw: pack all six weight matrices to bf16 (row-major preserved).
// Offsets (elements): wq 0, wk 16384, wv 32768, wmh 49152, wc1 65536, wc2 131072.
__global__ void pack_weights(const float* __restrict__ wq, const float* __restrict__ wk,
                             const float* __restrict__ wv, const float* __restrict__ wmh,
                             const float* __restrict__ wc1, const float* __restrict__ wc2,
                             __bf16* __restrict__ out) {
  int i = blockIdx.x * 256 + threadIdx.x;       // < 163840
  const float* src; int off;
  if      (i < 16384)  { src = wq;  off = 0; }
  else if (i < 32768)  { src = wk;  off = 16384; }
  else if (i < 49152)  { src = wv;  off = 32768; }
  else if (i < 65536)  { src = wmh; off = 49152; }
  else if (i < 131072) { src = wc1; off = 65536; }
  else                 { src = wc2; off = 131072; }
  out[i] = (__bf16)src[i - off];
}

// Kernel T: eigen [C][N] f32 -> catT [n][2C] bf16 (first C columns).
__global__ void pack_eigen(const float* __restrict__ eigen, __bf16* __restrict__ catT) {
  int idx = blockIdx.x * 256 + threadIdx.x;  // < N_*C_
  int n = idx >> 7, c = idx & 127;
  catT[n * (2 * C_) + c] = (__bf16)eigen[c * N_ + n];
}

// Kernel 1: QKV projections.  Q,K stored point-major [h][n][d] (for A / B-of-S
// fragments); V stored channel-major [c][n] (so P*V B-fragment is contiguous).
// The softmax logit scale 1/sqrt(HD) is folded into the Q store.
__global__ __launch_bounds__(32) void qkv_kernel(
    const __bf16* __restrict__ wpack,
    const float* __restrict__ bq, const float* __restrict__ bk,
    const float* __restrict__ bv, const __bf16* __restrict__ catT,
    __bf16* __restrict__ Qt, __bf16* __restrict__ Kt, __bf16* __restrict__ Vd) {
  int t = blockIdx.x;
  int mat = t >> 11;           // 0=q, 1=k, 2=v   (2048 tiles each)
  int rem = t & 2047;
  int co  = rem >> 8;          // 0..7
  int n0  = (rem & 255) * 16;
  const __bf16* W = wpack + mat * (C_ * C_);
  const float*  B = (mat == 0) ? bq : (mat == 1) ? bk : bv;
  floatx8 acc = gemm_tile(W, C_, catT, 2 * C_, co, n0);
  const int l = threadIdx.x & 31, lm = l & 15, hi = l >> 4;
  const float qscale = (mat == 0) ? 0.17677669529663687f : 1.f;  // 1/sqrt(32)
#pragma unroll
  for (int r = 0; r < 8; r++) {
    int c = co * 16 + r + hi * 8;   // D layout: VGPR r -> row r (lo) / r+8 (hi)
    int n = n0 + lm;
    float v = (acc[r] + B[c]) * qscale;
    if (mat == 2) {
      Vd[c * N_ + n] = (__bf16)v;                       // [h*32+d][n]
    } else {
      int h = c >> 5, d = c & 31;
      __bf16* dst = (mat == 0) ? Qt : Kt;
      dst[h * (N_ * HD_) + n * HD_ + d] = (__bf16)v;    // [h][n][d]
    }
  }
}

// Kernel 2: flash attention, one wave per (head, 16-query tile).
// S = (Q_tile[16x32] x K_chunk[32x16]) per 32 keys, online softmax with
// 16-lane butterfly reductions per accumulator row, P transposed to WMMA
// A-layout through a 1KB LDS tile, then two PV WMMAs (d 0..15 / 16..31).
__global__ __launch_bounds__(32) void fa_kernel(
    const __bf16* __restrict__ Qt, const __bf16* __restrict__ Kt,
    const __bf16* __restrict__ Vd, __bf16* __restrict__ AVt) {
  __shared__ __align__(64) __bf16 Pl[16 * 32];
  const int wid = blockIdx.x;
  const int h  = wid >> 8;
  const int n0 = (wid & 255) * 16;
  const int l = threadIdx.x & 31, lm = l & 15, hi = l >> 4;
  const __bf16* Qh = Qt + h * (N_ * HD_);
  const __bf16* Kh = Kt + h * (N_ * HD_);
  const __bf16* Vh = Vd + h * (HD_ * N_);

  const int ka = hi * 8;                        // A-layout K base for this lane
  const __bf16* qp = Qh + (n0 + lm) * HD_ + hi * 16;
  bf16x16 Aq = ld_bf8x2(qp, qp + 8);

  float mrow[8], lrow[8];
  floatx8 acc0 = {0.f, 0.f, 0.f, 0.f, 0.f, 0.f, 0.f, 0.f};
  floatx8 acc1 = {0.f, 0.f, 0.f, 0.f, 0.f, 0.f, 0.f, 0.f};
#pragma unroll
  for (int r = 0; r < 8; r++) { mrow[r] = -1e30f; lrow[r] = 0.f; }

  for (int m0 = 0; m0 < N_; m0 += 32) {
    const int dh = hi * 16;                     // B-layout K(d) base
    const __bf16* kp0 = Kh + (m0 + lm) * HD_ + dh;
    const __bf16* kp1 = Kh + (m0 + 16 + lm) * HD_ + dh;
    bf16x16 Bk0 = ld_bf8x2(kp0, kp0 + 8);
    bf16x16 Bk1 = ld_bf8x2(kp1, kp1 + 8);
    __builtin_prefetch(kp0 + 32 * HD_, 0, 0);   // global_prefetch_b8 next chunk
    floatx8 z = {0.f, 0.f, 0.f, 0.f, 0.f, 0.f, 0.f, 0.f};
    floatx8 S0 = wmma_bf16(Aq, Bk0, z);         // cols m0..m0+15 (pre-scaled Q)
    floatx8 S1 = wmma_bf16(Aq, Bk1, z);         // cols m0+16..m0+31

    float p0[8], p1[8];
#pragma unroll
    for (int r = 0; r < 8; r++) {               // row (r + 8*hi) of the tile
      float s0 = S0[r], s1 = S1[r];
      float mx = fmaxf(s0, s1);
#pragma unroll
      for (int m = 1; m < 16; m <<= 1) mx = fmaxf(mx, __shfl_xor(mx, m, 32));
      float mnew  = fmaxf(mrow[r], mx);
      float alpha = __expf(mrow[r] - mnew);
      float e0 = __expf(s0 - mnew), e1 = __expf(s1 - mnew);
      float sm = e0 + e1;
#pragma unroll
      for (int m = 1; m < 16; m <<= 1) sm += __shfl_xor(sm, m, 32);
      lrow[r] = lrow[r] * alpha + sm;
      mrow[r] = mnew;
      acc0[r] *= alpha; acc1[r] *= alpha;
      p0[r] = e0; p1[r] = e1;
    }
    // C-layout -> row-major P tile in LDS
#pragma unroll
    for (int r = 0; r < 8; r++) {
      int rr = r + hi * 8;
      Pl[rr * 32 + lm]      = (__bf16)p0[r];
      Pl[rr * 32 + 16 + lm] = (__bf16)p1[r];
    }
    asm volatile("s_wait_dscnt 0" ::: "memory");
    // Re-read P in WMMA A-layout (row = lm, K = m-offset)
    const __bf16* pp = &Pl[lm * 32 + hi * 16];
    bf16x16 Ap = ld_bf8x2(pp, pp + 8);
    // V B-fragments: K = key offset (contiguous in Vd rows), N = d
    const int mh2 = hi * 16;
    const __bf16* vp0 = Vh + lm * N_        + m0 + mh2;   // d = lm
    const __bf16* vp1 = Vh + (16 + lm) * N_ + m0 + mh2;   // d = 16+lm
    bf16x16 Bv0 = ld_bf8x2(vp0, vp0 + 8);
    bf16x16 Bv1 = ld_bf8x2(vp1, vp1 + 8);
    acc0 = wmma_bf16(Ap, Bv0, acc0);
    acc1 = wmma_bf16(Ap, Bv1, acc1);
  }
#pragma unroll
  for (int r = 0; r < 8; r++) {
    float inv = 1.f / lrow[r];
    int n = n0 + r + hi * 8;
    AVt[n * C_ + h * HD_ + lm]      = (__bf16)(acc0[r] * inv);
    AVt[n * C_ + h * HD_ + 16 + lm] = (__bf16)(acc1[r] * inv);
  }
}

// Kernel 3: multi-head output projection -> second half of catT.
__global__ __launch_bounds__(32) void mh_kernel(
    const __bf16* __restrict__ wmh16, const float* __restrict__ bmh,
    const __bf16* __restrict__ AVt, __bf16* __restrict__ catT) {
  int t = blockIdx.x;
  int co = t >> 8;
  int n0 = (t & 255) * 16;
  floatx8 acc = gemm_tile(wmh16, C_, AVt, C_, co, n0);
  const int l = threadIdx.x & 31, lm = l & 15, hi = l >> 4;
#pragma unroll
  for (int r = 0; r < 8; r++) {
    int c = co * 16 + r + hi * 8;
    int n = n0 + lm;
    catT[n * (2 * C_) + C_ + c] = (__bf16)(acc[r] + bmh[c]);
  }
}

// Kernel 4: h1 = wc1 @ cat + bc1, store f32 point-major, accumulate BN stats.
__global__ __launch_bounds__(32) void wc1_kernel(
    const __bf16* __restrict__ wc116, const float* __restrict__ bc1,
    const __bf16* __restrict__ catT,
    float* __restrict__ h1T, float* __restrict__ sum, float* __restrict__ sumsq) {
  int t = blockIdx.x;
  int co = t >> 8;               // 0..15
  int n0 = (t & 255) * 16;
  floatx8 acc = gemm_tile(wc116, 2 * C_, catT, 2 * C_, co, n0);
  const int l = threadIdx.x & 31, lm = l & 15, hi = l >> 4;
#pragma unroll
  for (int r = 0; r < 8; r++) {
    int c = co * 16 + r + hi * 8;
    int n = n0 + lm;
    float v = acc[r] + bc1[c];
    h1T[n * (2 * C_) + c] = v;
    float s = v, s2 = v * v;
#pragma unroll
    for (int m = 1; m < 16; m <<= 1) { s += __shfl_xor(s, m, 32); s2 += __shfl_xor(s2, m, 32); }
    if (lm == 0) { atomicAdd(&sum[c], s); atomicAdd(&sumsq[c], s2); }
  }
}

// Kernel 5: finalize BN: scale = gamma*rsqrt(var+eps), shift = beta - mean*scale.
__global__ void bn_finalize(const float* __restrict__ sum, const float* __restrict__ sumsq,
                            const float* __restrict__ gamma, const float* __restrict__ beta,
                            float* __restrict__ bnsc, float* __restrict__ bnsh) {
  int c = threadIdx.x;
  float mean = sum[c] * (1.f / N_);
  float var  = sumsq[c] * (1.f / N_) - mean * mean;
  float sc = gamma[c] * rsqrtf(var + 1e-5f);
  bnsc[c] = sc;
  bnsh[c] = beta[c] - mean * sc;
}

// Kernel 6: h2 = wc2 @ relu(BN(h1)) + bc2 + eigen (residual), BN+ReLU fused
// into the bf16 B-fragment conversion.  Output point-major f32.
__global__ __launch_bounds__(32) void wc2_kernel(
    const __bf16* __restrict__ wc216, const float* __restrict__ bc2,
    const float* __restrict__ h1T,
    const float* __restrict__ bnsc, const float* __restrict__ bnsh,
    const float* __restrict__ eigen, float* __restrict__ motT) {
  int t = blockIdx.x;
  int co = t >> 8;               // 0..7
  int n0 = (t & 255) * 16;
  const int l = threadIdx.x & 31, lm = l & 15, hi = l >> 4;
  const int rowA = co * 16 + lm;
  floatx8 acc = {0.f, 0.f, 0.f, 0.f, 0.f, 0.f, 0.f, 0.f};
  for (int c0 = 0; c0 < 2 * C_; c0 += 32) {
    int ka = c0 + hi * 16;
    const __bf16* ap = wc216 + rowA * (2 * C_) + ka;
    bf16x16 A = ld_bf8x2(ap, ap + 8);
    int kb = c0 + hi * 16;
    const float* hp = h1T + (n0 + lm) * (2 * C_) + kb;
    bf16x16 Bm;
#pragma unroll
    for (int jj = 0; jj < 16; jj++) {
      int c = kb + jj;
      float x = fmaf(hp[jj], bnsc[c], bnsh[c]);
      Bm[jj] = (__bf16)fmaxf(x, 0.f);
    }
    acc = wmma_bf16(A, Bm, acc);
  }
#pragma unroll
  for (int r = 0; r < 8; r++) {
    int c = co * 16 + r + hi * 8;
    int n = n0 + lm;
    motT[n * C_ + c] = acc[r] + bc2[c] + eigen[c * N_ + n];
  }
}

// Kernel 7: tran_layer  out[n][3] = wt[3x128] . motion[n] + bt.
__global__ void tran_kernel(const float* __restrict__ motT, const float* __restrict__ wt,
                            const float* __restrict__ bt, float* __restrict__ out) {
  int n = blockIdx.x * blockDim.x + threadIdx.x;
  if (n >= N_) return;
  const float* m = motT + n * C_;
  float a0 = bt[0], a1 = bt[1], a2 = bt[2];
#pragma unroll 4
  for (int c = 0; c < C_; c++) {
    float x = m[c];
    a0 = fmaf(wt[c], x, a0);
    a1 = fmaf(wt[C_ + c], x, a1);
    a2 = fmaf(wt[2 * C_ + c], x, a2);
  }
  out[n * 3 + 0] = a0; out[n * 3 + 1] = a1; out[n * 3 + 2] = a2;
}

// ---------------------------------------------------------------------------
extern "C" void kernel_launch(void* const* d_in, const int* in_sizes, int n_in,
                              void* d_out, int out_size, void* d_ws, size_t ws_size,
                              hipStream_t stream) {
  (void)in_sizes; (void)n_in; (void)out_size; (void)ws_size;
  const float* eigen = (const float*)d_in[0];
  const float* wq  = (const float*)d_in[1];  const float* bq  = (const float*)d_in[2];
  const float* wk  = (const float*)d_in[3];  const float* bk  = (const float*)d_in[4];
  const float* wv  = (const float*)d_in[5];  const float* bv  = (const float*)d_in[6];
  const float* wmh = (const float*)d_in[7];  const float* bmh = (const float*)d_in[8];
  const float* wc1 = (const float*)d_in[9];  const float* bc1 = (const float*)d_in[10];
  const float* gamma = (const float*)d_in[11]; const float* beta = (const float*)d_in[12];
  const float* wc2 = (const float*)d_in[13]; const float* bc2 = (const float*)d_in[14];
  const float* wt  = (const float*)d_in[15]; const float* bt  = (const float*)d_in[16];

  char* ws = (char*)d_ws;                        // ~12.4 MB used
  __bf16* catT = (__bf16*)(ws);                  // [4096][256] bf16  (2 MB)
  __bf16* Qt   = (__bf16*)(ws + (2u  << 20));    // [4][4096][32] bf16 (1 MB)
  __bf16* Kt   = (__bf16*)(ws + (3u  << 20));    // [4][4096][32] bf16 (1 MB)
  __bf16* Vd   = (__bf16*)(ws + (4u  << 20));    // [128][4096]  bf16 (1 MB)
  __bf16* AVt  = (__bf16*)(ws + (5u  << 20));    // [4096][128]  bf16 (1 MB)
  float*  h1T  = (float*)(ws + (6u  << 20));     // [4096][256]  f32  (4 MB)
  float*  motT = (float*)(ws + (10u << 20));     // [4096][128]  f32  (2 MB)
  float*  sum  = (float*)(ws + (12u << 20));     // 4 x 256 f32 stats
  float*  sumsq = sum + 256;
  float*  bnsc  = sum + 512;
  float*  bnsh  = sum + 768;
  __bf16* wpack = (__bf16*)(ws + (12u << 20) + 8192);  // 163840 bf16 (320 KB)

  zero_stats  <<<1, 512, 0, stream>>>(sum);
  pack_weights<<<163840 / 256, 256, 0, stream>>>(wq, wk, wv, wmh, wc1, wc2, wpack);
  pack_eigen  <<<(N_ * C_) / 256, 256, 0, stream>>>(eigen, catT);
  qkv_kernel  <<<3 * 8 * 256, 32, 0, stream>>>(wpack, bq, bk, bv, catT, Qt, Kt, Vd);
  fa_kernel   <<<H_ * (N_ / 16), 32, 0, stream>>>(Qt, Kt, Vd, AVt);
  mh_kernel   <<<8 * 256, 32, 0, stream>>>(wpack + 49152, bmh, AVt, catT);
  wc1_kernel  <<<16 * 256, 32, 0, stream>>>(wpack + 65536, bc1, catT, h1T, sum, sumsq);
  bn_finalize <<<1, 256, 0, stream>>>(sum, sumsq, gamma, beta, bnsc, bnsh);
  wc2_kernel  <<<8 * 256, 32, 0, stream>>>(wpack + 131072, bc2, h1T, bnsc, bnsh, eigen, motT);
  tran_kernel <<<N_ / 256, 256, 0, stream>>>(motT, wt, bt, (float*)d_out);
}
